// GraphConvLayer_54331336295029
// MI455X (gfx1250) — matrix operand order repaired and checked by reading, not compiled, not run on hardware
//
#include <hip/hip_runtime.h>

// ---------------------------------------------------------------------------
// GraphConv layer for gfx1250 (MI455X), bf16 WMMA path.
// B=4, N=128, D=512, H=8.
//
// Algebraic reduction: edge_input = [h_i | h_j | edge]  =>
//   edge_input @ W1 = (h@W1_a)[i] + (h@W1_b)[j] + edge @ W1_e
// so the per-edge GEMMs are DxD only (3 x 65536x512x512 ~ 103 GFLOP),
// done in bf16 WMMA with f32 accumulation.  Memory floor ~270MB @ 23.3TB/s.
// ---------------------------------------------------------------------------

typedef __attribute__((ext_vector_type(16))) __bf16 v16bf;
typedef __attribute__((ext_vector_type(8)))  __bf16 v8bf;
typedef __attribute__((ext_vector_type(4)))  __bf16 bf16x4;
typedef __attribute__((ext_vector_type(8)))  float  v8f;

#define LDSP 520   // LDS row pitch in bf16 elems (512 + 8 pad)

__device__ __forceinline__ float bf2f(__bf16 v) {
  unsigned short u = __builtin_bit_cast(unsigned short, v);
  unsigned int x = ((unsigned int)u) << 16;
  return __builtin_bit_cast(float, x);
}
__device__ __forceinline__ __bf16 f2bf(float f) {
  unsigned int u = __builtin_bit_cast(unsigned int, f);
  unsigned int r = u + 0x7FFFu + ((u >> 16) & 1u);  // round-to-nearest-even
  unsigned short h = (unsigned short)(r >> 16);
  return __builtin_bit_cast(__bf16, h);
}

// A-matrix 16x32 bf16 layout (ISA 7.12.2): lane L holds row M = L&15;
// elems 0..7  -> K = kbase + (L>>4)*8 + [0..7]
// elems 8..15 -> K = kbase + 16 + (L>>4)*8 + [0..7]
__device__ __forceinline__ v16bf ld_a(const __bf16* p) {
  v8bf lo = *(const v8bf*)(p);
  v8bf hi = *(const v8bf*)(p + 16);
  return __builtin_shufflevector(lo, hi, 0,1,2,3,4,5,6,7,8,9,10,11,12,13,14,15);
}

// B-matrix 32x16 bf16 layout: lane L holds col N = L&15; elem e is
// K = ktile*32 + (e>>3)*16 + (L>>4)*8 + (e&7).  Pre-packed so each lane's 16 elems are
// contiguous: offset = (((nt*16+kt)*32 + lane)*16).
__device__ __forceinline__ v16bf ld_b(const __bf16* __restrict__ pk, int nt, int kt, int lane) {
  const __bf16* p = pk + (((((nt << 4) + kt) << 5) + lane) << 4);
  v8bf lo = *(const v8bf*)(p);
  v8bf hi = *(const v8bf*)(p + 8);
  return __builtin_shufflevector(lo, hi, 0,1,2,3,4,5,6,7,8,9,10,11,12,13,14,15);
}

__device__ __forceinline__ v8f wmma_bf(v16bf a, v16bf b, v8f c) {
  return __builtin_amdgcn_wmma_f32_16x16x32_bf16(false, a, false, b, (short)0, c, false, false);
}

// One 16-row stripe x 8 col-tiles (128 cols) over the full K=512.
__device__ __forceinline__ void gemm_rowstripe8(const __bf16* __restrict__ Arow,
                                                const __bf16* __restrict__ Bpk,
                                                int lane, int cc, v8f* acc) {
  const int khalf = (lane >> 4) << 3;
  for (int kt = 0; kt < 16; ++kt) {
    v16bf Av = ld_a(Arow + kt * 32 + khalf);
#pragma unroll
    for (int q = 0; q < 8; ++q) {
      v16bf Bv = ld_b(Bpk, cc * 8 + q, kt, lane);
      acc[q] = wmma_bf(Av, Bv, acc[q]);
    }
  }
}

// ---------------------------------------------------------------------------
// pack_kernel: 9 weight matrices (512x512 f32 [K][N]) -> bf16 packed WMMA-B
// layout; node features -> bf16 row-major; plus a 512-float zero strip.
// ---------------------------------------------------------------------------
__global__ __launch_bounds__(256) void pack_kernel(
    const float* __restrict__ Wn, const float* __restrict__ We1,
    const float* __restrict__ Wm, const float* __restrict__ Wu1,
    const float* __restrict__ Wu2, const float* __restrict__ node,
    __bf16* __restrict__ pk, __bf16* __restrict__ nodebf,
    float* __restrict__ zeros) {
  int idx = blockIdx.x * 256 + threadIdx.x;
  if (idx >= 10 * 262144 + 512) return;
  int m = idx >> 18;
  int r = idx & 262143;
  if (m == 10) { zeros[r] = 0.f; return; }
  if (m == 9)  { nodebf[r] = f2bf(node[r]); return; }
  int elem = r & 15;
  int lane = (r >> 4) & 31;
  int kt   = (r >> 9) & 15;
  int nt   = r >> 13;
  int n = (nt << 4) + (lane & 15);
  int k = (kt << 5) + ((elem >> 3) << 4) + ((lane >> 4) << 3) + (elem & 7);
  float v;
  switch (m) {
    case 0: v = Wn[k * 512 + n]; break;
    case 1: v = We1[k * 512 + n]; break;               // We1_a
    case 2: v = We1[(k + 512) * 512 + n]; break;       // We1_b
    case 3: v = We1[(k + 1024) * 512 + n]; break;      // We1_e
    case 4: v = Wm[k * 512 + n]; break;
    case 5: v = Wu1[k * 512 + n]; break;               // Wu1_a
    case 6: v = Wu1[(k + 512) * 512 + n]; break;       // Wu1_b
    case 7: v = Wu1[(k + 1024) * 512 + n]; break;      // Wu1_e
    default: v = Wu2[k * 512 + n]; break;
  }
  pk[m * 262144 + r] = f2bf(v);
}

// ---------------------------------------------------------------------------
// Generic (512x512)@(512x512) bf16 WMMA GEMM + bias.  Branch-free epilogue:
// output type is a template parameter, bias is always a valid pointer.
// Grid = 4 blocks x 256 threads; block handles 128 rows (8 waves x 16).
// ---------------------------------------------------------------------------
template <bool BF16OUT>
__global__ __launch_bounds__(256) void gemm512_kernel(
    const __bf16* __restrict__ A, const __bf16* __restrict__ Bpk,
    const float* __restrict__ bias, void* __restrict__ outP) {
  const int tid = threadIdx.x, lane = tid & 31, wid = tid >> 5;
  const int lanelo = lane & 15, laneh = lane >> 4;
  const int rowbase = blockIdx.x * 128 + (wid << 4);
  const __bf16* Arow = A + (size_t)(rowbase + lanelo) * 512;
  for (int cc = 0; cc < 4; ++cc) {
    v8f acc[8];
    v8f z = {0.f, 0.f, 0.f, 0.f, 0.f, 0.f, 0.f, 0.f};
#pragma unroll
    for (int q = 0; q < 8; ++q) acc[q] = z;
    gemm_rowstripe8(Arow, Bpk, lane, cc, acc);
#pragma unroll
    for (int q = 0; q < 8; ++q) {
      int n = ((cc * 8 + q) << 4) + lanelo;
      float bv = bias[n];
#pragma unroll
      for (int r = 0; r < 8; ++r) {
        int mrow = rowbase + r + (laneh << 3);
        float v = acc[q][r] + bv;
        if constexpr (BF16OUT) ((__bf16*)outP)[(size_t)mrow * 512 + n] = f2bf(v);
        else                   ((float*)outP)[(size_t)mrow * 512 + n] = v;
      }
    }
  }
}

// ---------------------------------------------------------------------------
// edge_kernel: one workgroup per (b, i).  Fused:
//   e = leaky_relu(edge@We1_e + Pa'[i] + Pb[j]); logits = e@We2 + b;
//   softmax over j; * adj; mean heads; agg = attn_mean @ messages;
//   node_out = LN(node + agg@Wo + b);
//   u = gelu(edge@Wu1_e + Qa'[i] + Qb[j]); edge_out = edge + u@Wu2 + b.
// ---------------------------------------------------------------------------
__global__ __launch_bounds__(256) void edge_kernel(
    const float* __restrict__ edgeIn, const float* __restrict__ adjW,
    const float* __restrict__ nodeF,
    const float* __restrict__ PaP, const float* __restrict__ Pb,
    const float* __restrict__ QaP, const float* __restrict__ Qb,
    const float* __restrict__ Msg,
    const __bf16* __restrict__ pkWe1e, const __bf16* __restrict__ pkWu1e,
    const __bf16* __restrict__ pkWu2,
    const float* __restrict__ We2w, const float* __restrict__ We2b,
    const float* __restrict__ Wu2b,
    const float* __restrict__ Wow, const float* __restrict__ Wob,
    const float* __restrict__ lng, const float* __restrict__ lnb,
    float* __restrict__ outNodes, float* __restrict__ outEdges) {
  extern __shared__ char smem[];
  __bf16* bufE  = (__bf16*)(smem);                 // 128 x 520 bf16 = 133120 B
  __bf16* bufU  = (__bf16*)(smem + 133120);        // e, then u        133120 B
  float* we2s   = (float*)(smem + 266240);         // 512x8 f32        16384 B
  float* logits = (float*)(smem + 282624);         // 128x8 f32         4096 B
  float* adjs   = (float*)(smem + 286720);         // 128 f32            512 B
  float* am     = (float*)(smem + 287232);         // attn_mean          512 B
  float* agg    = (float*)(smem + 287744);         // 512 f32           2048 B
  float* ybuf   = (float*)(smem + 289792);         // 512 f32           2048 B
  float* red1   = (float*)(smem + 291840);         // 256 f32           1024 B
  float* red2   = (float*)(smem + 292864);         // 256 f32           1024 B
  float* hstat  = (float*)(smem + 293888);         // 16 f32              64 B

  const int tid = threadIdx.x, lane = tid & 31, wid = tid >> 5;
  const int lanelo = lane & 15, laneh = lane >> 4;
  const int bi = blockIdx.x;         // b*128 + i
  const int b  = bi >> 7;

  // ---- Phase 0: stage edge tile (128x512 f32 -> bf16), We2, adjacency ----
  const float* eptr = edgeIn + (size_t)bi * (128 * 512);
  for (int it = 0; it < 64; ++it) {
    int i4 = tid + it * 256;               // 0..16383 float4 chunks
    int l = i4 * 4;
    int row = l >> 9, k = l & 511;
    float4 v = *(const float4*)(eptr + l);
    bf16x4 p = { f2bf(v.x), f2bf(v.y), f2bf(v.z), f2bf(v.w) };
    *(bf16x4*)(bufE + row * LDSP + k) = p;
  }
  for (int idx = tid; idx < 4096; idx += 256) we2s[idx] = We2w[idx];
  if (tid < 128) adjs[tid] = adjW[(size_t)bi * 128 + tid];
  __syncthreads();

  const __bf16* ArowE = bufE + (size_t)((wid << 4) + lanelo) * LDSP;
  const __bf16* ArowU = bufU + (size_t)((wid << 4) + lanelo) * LDSP;
  v8f z = {0.f, 0.f, 0.f, 0.f, 0.f, 0.f, 0.f, 0.f};

  // ---- Phase 1: GEMM1: e = leaky_relu(edge@We1_e + Pa' + Pb[j]) -> bufU ----
  for (int cc = 0; cc < 4; ++cc) {
    v8f acc[8];
#pragma unroll
    for (int q = 0; q < 8; ++q) acc[q] = z;
    gemm_rowstripe8(ArowE, pkWe1e, lane, cc, acc);
#pragma unroll
    for (int q = 0; q < 8; ++q) {
      int n = ((cc * 8 + q) << 4) + lanelo;
      float pa = PaP[(size_t)bi * 512 + n];
#pragma unroll
      for (int r = 0; r < 8; ++r) {
        int jr = (wid << 4) + r + (laneh << 3);
        float v = acc[q][r] + pa + Pb[(size_t)(b * 128 + jr) * 512 + n];
        v = (v > 0.f) ? v : 0.2f * v;       // leaky_relu, slope 0.2
        bufU[jr * LDSP + n] = f2bf(v);
      }
    }
  }
  __syncthreads();

  // ---- Phase 2a: logits[j][h] = e[j,:] @ We2 + b ----
  {
    int j = tid >> 1;
    int hb = (tid & 1) * 4;
    float a0 = 0.f, a1 = 0.f, a2 = 0.f, a3 = 0.f;
    const __bf16* erow = bufU + (size_t)j * LDSP;
    for (int k = 0; k < 512; ++k) {
      float ev = bf2f(erow[k]);
      const float* wr = we2s + k * 8 + hb;
      a0 += ev * wr[0]; a1 += ev * wr[1]; a2 += ev * wr[2]; a3 += ev * wr[3];
    }
    logits[j * 8 + hb + 0] = a0 + We2b[hb + 0];
    logits[j * 8 + hb + 1] = a1 + We2b[hb + 1];
    logits[j * 8 + hb + 2] = a2 + We2b[hb + 2];
    logits[j * 8 + hb + 3] = a3 + We2b[hb + 3];
  }
  __syncthreads();
  // ---- Phase 2b: softmax stats over j, one wave per head (wave32 butterflies) ----
  {
    int h = wid;                            // 8 waves -> 8 heads
    float mx = -3.4e38f;
    for (int j = lane; j < 128; j += 32) mx = fmaxf(mx, logits[j * 8 + h]);
#pragma unroll
    for (int off = 16; off > 0; off >>= 1) mx = fmaxf(mx, __shfl_xor(mx, off, 32));
    float s = 0.f;
    for (int j = lane; j < 128; j += 32) s += expf(logits[j * 8 + h] - mx);
#pragma unroll
    for (int off = 16; off > 0; off >>= 1) s += __shfl_xor(s, off, 32);
    if (lane == 0) { hstat[h] = mx; hstat[8 + h] = 1.f / s; }
  }
  __syncthreads();
  if (tid < 128) {
    float s = 0.f;
    for (int h = 0; h < 8; ++h) s += expf(logits[tid * 8 + h] - hstat[h]) * hstat[8 + h];
    am[tid] = adjs[tid] * s * 0.125f;      // * adjacency, mean over heads
  }
  __syncthreads();
  // ---- Phase 2c: aggregated = attn_mean @ messages ----
  for (int d = tid; d < 512; d += 256) {
    float a = 0.f;
    const float* mcol = Msg + (size_t)(b * 128) * 512 + d;
    for (int j = 0; j < 128; ++j) a += am[j] * mcol[(size_t)j * 512];
    agg[d] = a;
  }
  __syncthreads();
  // ---- Phase 2d: node + agg@Wo + b, then LayerNorm ----
  for (int d2 = tid; d2 < 512; d2 += 256) {
    float a = Wob[d2];
    for (int d = 0; d < 512; ++d) a += agg[d] * Wow[(size_t)d * 512 + d2];
    ybuf[d2] = nodeF[(size_t)bi * 512 + d2] + a;
  }
  __syncthreads();
  {
    float y0 = ybuf[tid], y1 = ybuf[tid + 256];
    red1[tid] = y0 + y1;
    red2[tid] = y0 * y0 + y1 * y1;
  }
  __syncthreads();
  for (int off = 128; off > 0; off >>= 1) {
    if (tid < off) { red1[tid] += red1[tid + off]; red2[tid] += red2[tid + off]; }
    __syncthreads();
  }
  {
    float mean = red1[0] * (1.f / 512.f);
    float var = red2[0] * (1.f / 512.f) - mean * mean;
    float inv = rsqrtf(var + 1e-5f);
    for (int d2 = tid; d2 < 512; d2 += 256) {
      float y = ybuf[d2];
      outNodes[(size_t)bi * 512 + d2] = (y - mean) * inv * lng[d2] + lnb[d2];
    }
  }
  __syncthreads();

  // Warm the residual re-read (phase 4) back into WGP$/L2 while GEMM2 runs.
  for (int c = tid; c < 4096; c += 256)
    __builtin_prefetch((const char*)eptr + (size_t)c * 64, 0, 1);

  // ---- Phase 3: GEMM2: u = gelu(edge@Wu1_e + Qa' + Qb[j]) -> bufU ----
  for (int cc = 0; cc < 4; ++cc) {
    v8f acc[8];
#pragma unroll
    for (int q = 0; q < 8; ++q) acc[q] = z;
    gemm_rowstripe8(ArowE, pkWu1e, lane, cc, acc);
#pragma unroll
    for (int q = 0; q < 8; ++q) {
      int n = ((cc * 8 + q) << 4) + lanelo;
      float qa = QaP[(size_t)bi * 512 + n];
#pragma unroll
      for (int r = 0; r < 8; ++r) {
        int jr = (wid << 4) + r + (laneh << 3);
        float v = acc[q][r] + qa + Qb[(size_t)(b * 128 + jr) * 512 + n];
        v = 0.5f * v * (1.f + erff(v * 0.70710678118654752f));   // exact gelu
        bufU[jr * LDSP + n] = f2bf(v);
      }
    }
  }
  __syncthreads();

  // ---- Phase 4: GEMM3: edge_out = edge + u@Wu2 + b ----
  for (int cc = 0; cc < 4; ++cc) {
    v8f acc[8];
#pragma unroll
    for (int q = 0; q < 8; ++q) acc[q] = z;
    gemm_rowstripe8(ArowU, pkWu2, lane, cc, acc);
#pragma unroll
    for (int q = 0; q < 8; ++q) {
      int n = ((cc * 8 + q) << 4) + lanelo;
      float bb = Wu2b[n];
#pragma unroll
      for (int r = 0; r < 8; ++r) {
        int jr = (wid << 4) + r + (laneh << 3);
        size_t eo = ((size_t)bi * 128 + jr) * 512 + n;
        outEdges[eo] = edgeIn[eo] + acc[q][r] + bb;
      }
    }
  }
}

// ---------------------------------------------------------------------------
extern "C" void kernel_launch(void* const* d_in, const int* in_sizes, int n_in,
                              void* d_out, int out_size, void* d_ws, size_t ws_size,
                              hipStream_t stream) {
  (void)in_sizes; (void)n_in; (void)out_size; (void)ws_size;
  const float* node  = (const float*)d_in[0];
  const float* edge  = (const float*)d_in[1];
  const float* adj   = (const float*)d_in[2];
  const float* Wn_w  = (const float*)d_in[3];
  const float* Wn_b  = (const float*)d_in[4];
  const float* We1_w = (const float*)d_in[5];
  const float* We1_b = (const float*)d_in[6];
  const float* We2_w = (const float*)d_in[7];
  const float* We2_b = (const float*)d_in[8];
  const float* Wm_w  = (const float*)d_in[9];
  const float* Wm_b  = (const float*)d_in[10];
  const float* Wu1_w = (const float*)d_in[11];
  const float* Wu1_b = (const float*)d_in[12];
  const float* Wu2_w = (const float*)d_in[13];
  const float* Wu2_b = (const float*)d_in[14];
  const float* Wo_w  = (const float*)d_in[15];
  const float* Wo_b  = (const float*)d_in[16];
  const float* ln_g  = (const float*)d_in[17];
  const float* ln_b  = (const float*)d_in[18];

  char* ws = (char*)d_ws;
  __bf16* pk     = (__bf16*)(ws + 0);         // 9 x 512KB packed bf16 weights
  __bf16* nodebf = (__bf16*)(ws + 4718592);
  __bf16* hbf    = (__bf16*)(ws + 5242880);
  float* Pa    = (float*)(ws + 5767168);      // h@We1_a + be1
  float* Pb    = (float*)(ws + 6815744);      // h@We1_b
  float* Qa    = (float*)(ws + 7864320);      // h@Wu1_a + bu1
  float* Qb    = (float*)(ws + 8912896);      // h@Wu1_b
  float* Msg   = (float*)(ws + 9961472);      // h@Wm + bm
  float* zeros = (float*)(ws + 11010048);     // 512 zero floats (bias stub)

  float* outNodes = (float*)d_out;
  float* outEdges = outNodes + 262144;

  pack_kernel<<<10242, 256, 0, stream>>>(Wn_w, We1_w, Wm_w, Wu1_w, Wu2_w, node,
                                         pk, nodebf, zeros);
  // h = node@Wn + bn  (bf16 out)
  gemm512_kernel<true><<<4, 256, 0, stream>>>(nodebf, pk + 0 * 262144, Wn_b, hbf);
  // node-side precomputations (f32 out)
  gemm512_kernel<false><<<4, 256, 0, stream>>>(hbf, pk + 1 * 262144, We1_b, Pa);
  gemm512_kernel<false><<<4, 256, 0, stream>>>(hbf, pk + 2 * 262144, zeros, Pb);
  gemm512_kernel<false><<<4, 256, 0, stream>>>(hbf, pk + 5 * 262144, Wu1_b, Qa);
  gemm512_kernel<false><<<4, 256, 0, stream>>>(hbf, pk + 6 * 262144, zeros, Qb);
  gemm512_kernel<false><<<4, 256, 0, stream>>>(hbf, pk + 4 * 262144, Wm_b, Msg);

  // fused edge pipeline: one workgroup per (b,i); ~287KB dynamic LDS
  edge_kernel<<<512, 256, 293952, stream>>>(
      edge, adj, node, Pa, Pb, Qa, Qb, Msg,
      pk + 3 * 262144, pk + 7 * 262144, pk + 8 * 262144,
      We2_w, We2_b, Wu2_b, Wo_w, Wo_b, ln_g, ln_b, outNodes, outEdges);
}
